// MyLSTM_12549894438929
// MI455X (gfx1250) — matrix operand order repaired and checked by reading, not compiled, not run on hardware
//
#include <hip/hip_runtime.h>
#include <hip/hip_bf16.h>

// ---------------------------------------------------------------------------
// LSTM (2 layers, S=512, B=64, D=H=1024) for gfx1250 (MI455X).
//   Phase 1 per layer: x_proj = x @ Wih^T + (bih+bhh)   -- big bf16 WMMA GEMM
//     (compute-bound, AI ~420 FLOP/B -> bf16 16x16x32 WMMA, f32 accumulate)
//   Phase 2 per layer: 512 sequential fused step kernels. Whh panel for the
//     block's hidden-tile (4 gates x 16 rows x K=1024 = 128KB) is staged into
//     LDS by the Tensor Data Mover (one 3D-tile descriptor), then all 4 waves
//     run WMMA from LDS; gate activations + c/h update happen in registers.
//   GEMM loops use ping-pong double buffering with a 2x-unrolled steady state
//   (no loop-carried register copies) so loads overlap the matrix pipe.
// ---------------------------------------------------------------------------

typedef __bf16 bf16;
typedef __attribute__((ext_vector_type(16))) __bf16 v16bf;
typedef __attribute__((ext_vector_type(8)))  __bf16 v8bf;
typedef __attribute__((ext_vector_type(8)))  float  v8f;
typedef __attribute__((ext_vector_type(4)))  unsigned v4u;
typedef __attribute__((ext_vector_type(8)))  int      v8i;
typedef __attribute__((ext_vector_type(4)))  int      v4i;

#define S_LEN 512
#define B_SZ  64
#define H_SZ  1024
#define G_SZ  4096   // 4*H
#define L_SZ  2
#define M_ALL 32768  // S*B

static __device__ __forceinline__ v16bf cat16(v8bf lo, v8bf hi) {
    return __builtin_shufflevector(lo, hi, 0,1,2,3,4,5,6,7,8,9,10,11,12,13,14,15);
}

static __device__ __forceinline__ v16bf load_frag_a(const bf16* p) {
    return cat16(*(const v8bf*)p, *(const v8bf*)(p + 8));
}
static __device__ __forceinline__ v16bf load_frag_b(const bf16* p) {
    return cat16(*(const v8bf*)p, *(const v8bf*)(p + 8));
}

static __device__ __forceinline__ float fsig(float x) {
    return 1.0f / (1.0f + __expf(-x));
}

// ---------------------------------------------------------------------------
// f32 -> bf16 conversion / zero fill
// ---------------------------------------------------------------------------
__global__ void cvt_f32_bf16_kernel(const float* __restrict__ in,
                                    bf16* __restrict__ out, int n) {
    int i = blockIdx.x * blockDim.x + threadIdx.x;
    if (i < n) out[i] = (bf16)in[i];
}

__global__ void zero_u32_kernel(unsigned* __restrict__ p, int n) {
    int i = blockIdx.x * blockDim.x + threadIdx.x;
    if (i < n) p[i] = 0u;
}

// ---------------------------------------------------------------------------
// x_proj GEMM: out[M,4096] = x[M,1024](bf16) @ wih[4096,1024]^T (bf16) + bias
// One wave -> 16(M) x 64(N) tile (4 accumulators). 4 waves / block.
// Ping-pong pipelined: buffers alternate roles, no register copies.
// ---------------------------------------------------------------------------
__global__ void __launch_bounds__(128, 1)
lstm_xproj_kernel(const bf16* __restrict__ x,
                  const bf16* __restrict__ wih,
                  const float* __restrict__ bih,
                  const float* __restrict__ bhh,
                  float* __restrict__ out) {
    const int lane = threadIdx.x & 31;
    const int wid  = blockIdx.x * 4 + (threadIdx.x >> 5);
    const int n64  = wid & 63;        // 64-wide N tile index (N=4096)
    const int m0   = (wid >> 6) * 16; // M tile base
    const int half = lane >> 4;
    const int r    = lane & 15;

    v8f acc[4];
    #pragma unroll
    for (int j = 0; j < 4; ++j) {
        const int col = n64 * 64 + j * 16 + r;
        const float bias = bih[col] + bhh[col];
        #pragma unroll
        for (int e = 0; e < 8; ++e) acc[j][e] = bias;
    }

    const bf16* arow = x   + (m0 + r) * H_SZ + 16 * half;        // A frag base
    const bf16* brow = wih + (n64 * 64 + r) * H_SZ + 16 * half;  // B frag base

    v16bf a0, a1, b0[4], b1[4];
    a0 = load_frag_a(arow);
    #pragma unroll
    for (int j = 0; j < 4; ++j) b0[j] = load_frag_b(brow + j * 16 * H_SZ);

    for (int k0 = 0; k0 < H_SZ - 64; k0 += 64) {
        // phase 1: fetch k0+32 into buf1, compute buf0 (k0)
        a1 = load_frag_a(arow + k0 + 32);
        #pragma unroll
        for (int j = 0; j < 4; ++j)
            b1[j] = load_frag_b(brow + j * 16 * H_SZ + k0 + 32);
        __builtin_prefetch(arow + k0 + 64, 0, 3);   // global_prefetch_b8
        #pragma unroll
        for (int j = 0; j < 4; ++j)
            acc[j] = __builtin_amdgcn_wmma_f32_16x16x32_bf16(
                false, a0, false, b0[j], (short)0, acc[j], false, false);
        // phase 2: fetch k0+64 into buf0, compute buf1 (k0+32)
        a0 = load_frag_a(arow + k0 + 64);
        #pragma unroll
        for (int j = 0; j < 4; ++j)
            b0[j] = load_frag_b(brow + j * 16 * H_SZ + k0 + 64);
        #pragma unroll
        for (int j = 0; j < 4; ++j)
            acc[j] = __builtin_amdgcn_wmma_f32_16x16x32_bf16(
                false, a1, false, b1[j], (short)0, acc[j], false, false);
    }
    // epilogue: buf0 holds k = H-64; fetch/compute last chunk k = H-32
    a1 = load_frag_a(arow + H_SZ - 32);
    #pragma unroll
    for (int j = 0; j < 4; ++j)
        b1[j] = load_frag_b(brow + j * 16 * H_SZ + H_SZ - 32);
    #pragma unroll
    for (int j = 0; j < 4; ++j)
        acc[j] = __builtin_amdgcn_wmma_f32_16x16x32_bf16(
            false, a0, false, b0[j], (short)0, acc[j], false, false);
    #pragma unroll
    for (int j = 0; j < 4; ++j)
        acc[j] = __builtin_amdgcn_wmma_f32_16x16x32_bf16(
            false, a1, false, b1[j], (short)0, acc[j], false, false);

    #pragma unroll
    for (int j = 0; j < 4; ++j) {
        const int col = n64 * 64 + j * 16 + r;
        #pragma unroll
        for (int e = 0; e < 8; ++e) {
            const int m = m0 + e + 8 * half;
            out[m * G_SZ + col] = acc[j][e];
        }
    }
}

// ---------------------------------------------------------------------------
// Fused recurrent step.
// Block = 4 waves (128 thr). n-tile is uniform per block (n0 = blockIdx.x*16),
// waves take the 4 batch tiles. The block's Whh panel:
//   rows {g*1024 + n0 .. +15}, g=0..3, all K  => 3D tile 1024 x 16 x 4 (bf16)
// is DMA'd into LDS by the TDM once, then the K-loop reads B from LDS.
// ---------------------------------------------------------------------------
__global__ void __launch_bounds__(128, 1)
lstm_step_kernel(const float* __restrict__ xp,      // [B,4H] this timestep
                 const bf16*  __restrict__ whh,     // [4H,H] bf16
                 const bf16*  __restrict__ h_in,    // [B,H] bf16
                 bf16*        __restrict__ h_out_bf,// [B,H]
                 float*       __restrict__ c_state, // [B,H]
                 float*       __restrict__ h_state, // [B,H]
                 bf16*        __restrict__ seq_bf,  // layer0 out (or null)
                 float*       __restrict__ seq_f32) // layer1 out (or null)
{
    __shared__ bf16 ldsB[4 * 16 * H_SZ];            // 128 KB

    const int lane = threadIdx.x & 31;
    const int widx = threadIdx.x >> 5;
    const int m0   = widx * 16;                     // batch tile (B=64)
    const int n0   = blockIdx.x * 16;               // hidden tile (uniform)
    const int half = lane >> 4;
    const int r    = lane & 15;
    const int col  = n0 + r;

    // --- TDM: stage Whh panel into LDS (one wave issues; EXEC is ignored) ---
    if (widx == 0) {
        const unsigned lds_base = (unsigned)(unsigned long long)(&ldsB[0]);
        const unsigned long long gaddr =
            (unsigned long long)(const void*)(whh + (size_t)n0 * H_SZ);
        // D# group0: count=1, lds_addr, 57-bit global_addr, type=2
        v4u g0;
        g0[0] = 1u;
        g0[1] = lds_base;
        g0[2] = (unsigned)(gaddr & 0xffffffffull);
        g0[3] = (unsigned)((gaddr >> 32) & 0x1ffffffull) | (2u << 30);
        // D# group1: data_size=2B; tensor_dim0=1024, tensor_dim1=4096;
        // tile 1024 x 16 x 4; dim0 stride 1024, dim1 stride 1024*1024 (elems)
        v8i g1;
        g1[0] = (int)(1u << 16);                      // data_size=1 (2 bytes)
        g1[1] = (int)((1024u & 0xffffu) << 16);       // tensor_dim0 lo16
        g1[2] = (int)((1024u >> 16) | ((4096u & 0xffffu) << 16)); // d0 hi | d1 lo
        g1[3] = (int)((4096u >> 16) | (1024u << 16)); // d1 hi | tile_dim0=1024
        g1[4] = (int)(16u | (4u << 16));              // tile_dim1=16, tile_dim2=4
        g1[5] = (int)1024u;                           // tensor_dim0_stride lo32
        g1[6] = (int)(((1024u * 1024u) & 0xffffu) << 16); // stride hi | d1stride lo16
        g1[7] = (int)((1024u * 1024u) >> 16);         // tensor_dim1_stride[47:16]
        // D# group2: tensor_dim2=4 (3D tile), rest zero; group3 + extra: zero
        v4i g2; g2[0] = 4; g2[1] = 0; g2[2] = 0; g2[3] = 0;
        v4i g3; g3[0] = 0; g3[1] = 0; g3[2] = 0; g3[3] = 0;
        v8i g4;
        #pragma unroll
        for (int i = 0; i < 8; ++i) g4[i] = 0;
        __builtin_amdgcn_tensor_load_to_lds(g0, g1, g2, g3, g4, 0);
        __builtin_amdgcn_s_wait_tensorcnt(0);
    }
    __syncthreads();

    // --- init accumulators from x_proj (gate blocks i,f,g,o) ---
    v8f acc[4];
    #pragma unroll
    for (int g = 0; g < 4; ++g) {
        #pragma unroll
        for (int e = 0; e < 8; ++e) {
            const int m = m0 + e + 8 * half;
            acc[g][e] = xp[m * G_SZ + g * H_SZ + col];
        }
    }

    // --- gates += h_in @ whh^T; A from global, B from LDS (ping-pong) ---
    const bf16* arow = h_in + (m0 + r) * H_SZ + 16 * half;
    const bf16* lrow = &ldsB[r * H_SZ + 16 * half];

    v16bf a0, a1, b0[4], b1[4];
    a0 = load_frag_a(arow);
    #pragma unroll
    for (int g = 0; g < 4; ++g) b0[g] = load_frag_b(lrow + g * 16 * H_SZ);

    for (int k0 = 0; k0 < H_SZ - 64; k0 += 64) {
        a1 = load_frag_a(arow + k0 + 32);
        #pragma unroll
        for (int g = 0; g < 4; ++g)
            b1[g] = load_frag_b(lrow + g * 16 * H_SZ + k0 + 32);
        #pragma unroll
        for (int g = 0; g < 4; ++g)
            acc[g] = __builtin_amdgcn_wmma_f32_16x16x32_bf16(
                false, a0, false, b0[g], (short)0, acc[g], false, false);
        a0 = load_frag_a(arow + k0 + 64);
        #pragma unroll
        for (int g = 0; g < 4; ++g)
            b0[g] = load_frag_b(lrow + g * 16 * H_SZ + k0 + 64);
        #pragma unroll
        for (int g = 0; g < 4; ++g)
            acc[g] = __builtin_amdgcn_wmma_f32_16x16x32_bf16(
                false, a1, false, b1[g], (short)0, acc[g], false, false);
    }
    a1 = load_frag_a(arow + H_SZ - 32);
    #pragma unroll
    for (int g = 0; g < 4; ++g)
        b1[g] = load_frag_b(lrow + g * 16 * H_SZ + H_SZ - 32);
    #pragma unroll
    for (int g = 0; g < 4; ++g)
        acc[g] = __builtin_amdgcn_wmma_f32_16x16x32_bf16(
            false, a0, false, b0[g], (short)0, acc[g], false, false);
    #pragma unroll
    for (int g = 0; g < 4; ++g)
        acc[g] = __builtin_amdgcn_wmma_f32_16x16x32_bf16(
            false, a1, false, b1[g], (short)0, acc[g], false, false);

    // --- cell update in registers ---
    #pragma unroll
    for (int e = 0; e < 8; ++e) {
        const int m   = m0 + e + 8 * half;
        const int idx = m * H_SZ + col;
        const float iv = fsig(acc[0][e]);
        const float fv = fsig(acc[1][e]);
        const float gv = tanhf(acc[2][e]);
        const float ov = fsig(acc[3][e]);
        const float c  = fv * c_state[idx] + iv * gv;
        const float h  = ov * tanhf(c);
        c_state[idx]  = c;
        h_state[idx]  = h;
        const bf16 hb = (bf16)h;
        h_out_bf[idx] = hb;
        if (seq_bf)  seq_bf[idx]  = hb;
        if (seq_f32) seq_f32[idx] = h;
    }
}

// ---------------------------------------------------------------------------
// Host-side orchestration (graph-capture safe).
// ---------------------------------------------------------------------------
extern "C" void kernel_launch(void* const* d_in, const int* in_sizes, int n_in,
                              void* d_out, int out_size, void* d_ws, size_t ws_size,
                              hipStream_t stream) {
    const float* X    = (const float*)d_in[0]; // [S,B,D]
    const float* W_ih = (const float*)d_in[1]; // [L,4H,D]
    const float* W_hh = (const float*)d_in[2]; // [L,4H,H]
    const float* b_ih = (const float*)d_in[3]; // [L,4H]
    const float* b_hh = (const float*)d_in[4]; // [L,4H]
    float* out = (float*)d_out;                // output[S,B,H] ++ h_n[L,B,H] ++ c_n[L,B,H]

    char* w = (char*)d_ws;
    float* xproj   = (float*)(w);                                  // 512 MB [S*B,4H] f32
    bf16*  xbf     = (bf16*) (w + 536870912ULL);                   //  64 MB X bf16
    bf16*  hseq    = (bf16*) (w + 603979776ULL);                   //  64 MB layer0 h-seq bf16
    bf16*  wihbf   = (bf16*) (w + 671088640ULL);                   //  16 MB [L,4H,D] bf16
    bf16*  whhbf   = (bf16*) (w + 687865856ULL);                   //  16 MB [L,4H,H] bf16
    float* h_state = (float*)(w + 704643072ULL);                   // 512 KB [L,B,H]
    float* c_state = (float*)(w + 705167360ULL);                   // 512 KB [L,B,H]
    bf16*  hbf     = (bf16*) (w + 705691648ULL);                   // 256 KB [2,B,H]

    const int BHW = B_SZ * H_SZ;          // 65536
    const int WDI = G_SZ * H_SZ;          // 4M elems per layer weight

    {   // X -> bf16
        const int n = M_ALL * H_SZ;
        cvt_f32_bf16_kernel<<<(n + 255) / 256, 256, 0, stream>>>(X, xbf, n);
    }

    for (int l = 0; l < L_SZ; ++l) {
        bf16* wih_l = wihbf + (size_t)l * WDI;
        bf16* whh_l = whhbf + (size_t)l * WDI;
        cvt_f32_bf16_kernel<<<(WDI + 255) / 256, 256, 0, stream>>>(
            W_ih + (size_t)l * WDI, wih_l, WDI);
        cvt_f32_bf16_kernel<<<(WDI + 255) / 256, 256, 0, stream>>>(
            W_hh + (size_t)l * WDI, whh_l, WDI);

        const bf16* x_l = (l == 0) ? xbf : hseq;
        lstm_xproj_kernel<<<(2048 * 64) / 4, 128, 0, stream>>>(
            x_l, wih_l, b_ih + (size_t)l * G_SZ, b_hh + (size_t)l * G_SZ, xproj);

        // zero c state and both h double-buffers
        zero_u32_kernel<<<(BHW + 255) / 256, 256, 0, stream>>>(
            (unsigned*)(c_state + (size_t)l * BHW), BHW);
        zero_u32_kernel<<<(BHW + 255) / 256, 256, 0, stream>>>(
            (unsigned*)hbf, BHW);   // 2*BHW bf16 == BHW u32 words

        for (int t = 0; t < S_LEN; ++t) {
            lstm_step_kernel<<<64, 128, 0, stream>>>(
                xproj + (size_t)t * B_SZ * G_SZ,
                whh_l,
                hbf + (size_t)(t & 1) * BHW,
                hbf + (size_t)((t + 1) & 1) * BHW,
                c_state + (size_t)l * BHW,
                h_state + (size_t)l * BHW,
                (l == 0) ? (hseq + (size_t)t * BHW) : (bf16*)nullptr,
                (l == 1) ? (out + (size_t)t * BHW) : (float*)nullptr);
        }
    }

    float* h_dst = out + (size_t)M_ALL * H_SZ;
    float* c_dst = h_dst + (size_t)L_SZ * BHW;
    (void)hipMemcpyAsync(h_dst, h_state, (size_t)L_SZ * BHW * sizeof(float),
                         hipMemcpyDeviceToDevice, stream);
    (void)hipMemcpyAsync(c_dst, c_state, (size_t)L_SZ * BHW * sizeof(float),
                         hipMemcpyDeviceToDevice, stream);
}
